// Floodfill_81801947120387
// MI455X (gfx1250) — matrix operand order repaired and checked by reading, not compiled, not run on hardware
//
#include <hip/hip_runtime.h>
#include <hip/hip_bf16.h>
#include <math.h>

// ---------------- constants from the reference ----------------
#define NUM_ADJ   9
#define BW        512
#define RADIUS_C  0.1f
#define FSTP      120
#define INLIER_C  0.1f
#define LGR       5
#define TRI_THR   0.1f
#define NMS_RANGE 0.05f
#define FILL_X    20
#define KCOV      24
#define LL        123               // 3 + FSTP*NPS
#define EPS_C     1e-12f
#define BIGF      1000000.0f

#define CAP    32768                // candidate pool (power of 2 for bitonic)
#define TOPN   (BW*FILL_X)          // 10240
#define NBINS  4096
#define NSTAGE (LL*NUM_ADJ)         // 1107

typedef __bf16 bf16_t;
typedef bf16_t v16bf __attribute__((ext_vector_type(16)));
typedef bf16_t v8bf  __attribute__((ext_vector_type(8)));
typedef float  v8f   __attribute__((ext_vector_type(8)));

// ---------------- small helpers ----------------
__device__ __forceinline__ void argmin128(float v, int i, float* rv, int* ri, int tid) {
  __syncthreads();
  rv[tid] = v; ri[tid] = i;
  __syncthreads();
  for (int s = 64; s > 0; s >>= 1) {
    if (tid < s) {
      float a = rv[tid], b = rv[tid + s];
      int ai = ri[tid], bi = ri[tid + s];
      if (b < a || (b == a && bi < ai)) { rv[tid] = b; ri[tid] = bi; }
    }
    __syncthreads();
  }
}

__device__ __forceinline__ float dist3(const float* p, float x, float y, float z) {
  float dx = p[0] - x, dy = p[1] - y, dz = p[2] - z;
  return sqrtf(dx * dx + dy * dy + dz * dz + EPS_C);
}

// ---------------- fp32 -> bf16 ----------------
__global__ void cvt_bf16_kernel(const float* __restrict__ in, bf16_t* __restrict__ out, int n) {
  int i = blockIdx.x * blockDim.x + threadIdx.x;
  if (i < n) out[i] = (bf16_t)in[i];
}

// ---------------- WMMA GEMM: logits = A @ B^T * scale ----------------
// Register-blocked: one wave computes a 16(M) x 64(N) strip with 4 accumulators,
// reusing the A fragment 4x per K-chunk (load:wmma = 2.5 vs 4 unblocked).
// A-operand layout (16-bit A 16x32, this kernel's contiguous k order, not the ISA table order): lane = 16*half + m; elems e -> K = half*16 + e.
// B-operand layout (16-bit B 32x16, same contiguous k order as A, not the ISA table order): lane = 16*half + n; elems e -> K = half*16 + e.
__global__ void wmma_logits_kernel(const bf16_t* __restrict__ A, const bf16_t* __restrict__ B,
                                   float* __restrict__ C, int N, int K, float scale) {
  int lane = threadIdx.x;
  int half = lane >> 4, r = lane & 15;
  int tm  = blockIdx.y * blockDim.y + threadIdx.y;  // M tile (16 rows)
  int tn0 = blockIdx.x * 4;                         // first of 4 N tiles
  const bf16_t* arow = A + (size_t)(tm * 16 + r) * K;
  const bf16_t* brow0 = B + (size_t)(tn0 * 16 + r) * K;
  const bf16_t* brow1 = brow0 + (size_t)16 * K;
  const bf16_t* brow2 = brow0 + (size_t)32 * K;
  const bf16_t* brow3 = brow0 + (size_t)48 * K;
  v8f acc0 = {}, acc1 = {}, acc2 = {}, acc3 = {};
  for (int kk = 0; kk < K; kk += 32) {
    __builtin_prefetch(arow + kk + 64, 0, 1);
    __builtin_prefetch(brow0 + kk + 64, 0, 1);
    __builtin_prefetch(brow1 + kk + 64, 0, 1);
    __builtin_prefetch(brow2 + kk + 64, 0, 1);
    __builtin_prefetch(brow3 + kk + 64, 0, 1);
    union { v16bf v; v8bf h[2]; } a, b0, b1, b2, b3;
    a.h[0]  = *(const v8bf*)(arow + kk + half * 16);
    a.h[1]  = *(const v8bf*)(arow + kk + half * 16 + 8);
    b0.h[0] = *(const v8bf*)(brow0 + kk + half * 16);
    b0.h[1] = *(const v8bf*)(brow0 + kk + half * 16 + 8);
    b1.h[0] = *(const v8bf*)(brow1 + kk + half * 16);
    b1.h[1] = *(const v8bf*)(brow1 + kk + half * 16 + 8);
    b2.h[0] = *(const v8bf*)(brow2 + kk + half * 16);
    b2.h[1] = *(const v8bf*)(brow2 + kk + half * 16 + 8);
    b3.h[0] = *(const v8bf*)(brow3 + kk + half * 16);
    b3.h[1] = *(const v8bf*)(brow3 + kk + half * 16 + 8);
    acc0 = __builtin_amdgcn_wmma_f32_16x16x32_bf16(false, a.v, false, b0.v, (short)0, acc0,
                                                   false, false);
    acc1 = __builtin_amdgcn_wmma_f32_16x16x32_bf16(false, a.v, false, b1.v, (short)0, acc1,
                                                   false, false);
    acc2 = __builtin_amdgcn_wmma_f32_16x16x32_bf16(false, a.v, false, b2.v, (short)0, acc2,
                                                   false, false);
    acc3 = __builtin_amdgcn_wmma_f32_16x16x32_bf16(false, a.v, false, b3.v, (short)0, acc3,
                                                   false, false);
  }
  // C layout: VGPR v holds M = v + 8*half, N = r (within each 16x16 tile)
  float* cp = C + (size_t)(tm * 16 + half * 8) * N + tn0 * 16 + r;
#pragma unroll
  for (int v = 0; v < 8; ++v) {
    cp[(size_t)v * N +  0] = acc0[v] * scale;
    cp[(size_t)v * N + 16] = acc1[v] * scale;
    cp[(size_t)v * N + 32] = acc2[v] * scale;
    cp[(size_t)v * N + 48] = acc3[v] * scale;
  }
}

// ---------------- softmax stats ----------------
__global__ void row_stats_kernel(const float* __restrict__ L, float* rmax, float* rsum, int nc) {
  __shared__ float red[256];
  int i = blockIdx.x, tid = threadIdx.x;
  const float* row = L + (size_t)i * nc;
  float m = -INFINITY;
  for (int j = tid; j < nc; j += 256) m = fmaxf(m, row[j]);
  red[tid] = m; __syncthreads();
  for (int s = 128; s > 0; s >>= 1) { if (tid < s) red[tid] = fmaxf(red[tid], red[tid + s]); __syncthreads(); }
  m = red[0]; __syncthreads();
  float sm = 0.f;
  for (int j = tid; j < nc; j += 256) sm += expf(row[j] - m);
  red[tid] = sm; __syncthreads();
  for (int s = 128; s > 0; s >>= 1) { if (tid < s) red[tid] += red[tid + s]; __syncthreads(); }
  if (tid == 0) { rmax[i] = m; rsum[i] = red[0]; }
}

__global__ void col_stats_kernel(const float* __restrict__ L, float* cmax, float* csum,
                                 int nr_rows, int nc) {
  __shared__ float red[256];
  int j = blockIdx.x, tid = threadIdx.x;
  float m = -INFINITY;
  for (int i = tid; i < nr_rows; i += 256) m = fmaxf(m, L[(size_t)i * nc + j]);
  red[tid] = m; __syncthreads();
  for (int s = 128; s > 0; s >>= 1) { if (tid < s) red[tid] = fmaxf(red[tid], red[tid + s]); __syncthreads(); }
  m = red[0]; __syncthreads();
  float sm = 0.f;
  for (int i = tid; i < nr_rows; i += 256) sm += expf(L[(size_t)i * nc + j] - m);
  red[tid] = sm; __syncthreads();
  for (int s = 128; s > 0; s >>= 1) { if (tid < s) red[tid] += red[tid + s]; __syncthreads(); }
  if (tid == 0) { cmax[j] = m; csum[j] = red[0]; }
}

// ---------------- score (in place over logits) + global max ----------------
__global__ void score_kernel(float* __restrict__ L, const float* rmax, const float* rsum,
                             const float* cmax, const float* csum,
                             const float* es, const float* er,
                             int n_s, int n_r, int* gmax) {
  __shared__ float red[256];
  size_t total = (size_t)n_s * n_r;
  size_t stride = (size_t)gridDim.x * blockDim.x;
  float lm = 0.f;
  for (size_t t = (size_t)blockIdx.x * blockDim.x + threadIdx.x; t < total; t += stride) {
    int i = (int)(t / n_r), j = (int)(t % n_r);
    float l = L[t];
    float s = (expf(l - rmax[i]) / rsum[i]) * (expf(l - cmax[j]) / csum[j]) * es[i] * er[j];
    L[t] = s;
    lm = fmaxf(lm, s);
  }
  int tid = threadIdx.x;
  red[tid] = lm; __syncthreads();
  for (int s = 128; s > 0; s >>= 1) { if (tid < s) red[tid] = fmaxf(red[tid], red[tid + s]); __syncthreads(); }
  if (tid == 0) atomicMax(gmax, __float_as_int(red[0]));   // non-negative floats: int order == float order
}

// ---------------- init scratch control state ----------------
__global__ void init_kernel(int* gmax, int* ncand, int* hist, float* candS, int* candI) {
  int t = blockIdx.x * blockDim.x + threadIdx.x;
  if (t == 0) { *gmax = 0; *ncand = 0; }
  if (t < NBINS) hist[t] = 0;
  if (t < CAP) { candS[t] = -1.0f; candI[t] = 0; }
}

// ---------------- histogram / threshold / compact ----------------
__global__ void hist_kernel(const float* __restrict__ S, size_t total, const int* gmaxi, int* hist) {
  __shared__ int h[NBINS];
  for (int i = threadIdx.x; i < NBINS; i += blockDim.x) h[i] = 0;
  __syncthreads();
  float gm = __int_as_float(*gmaxi);
  float sc = (gm > 0.f) ? ((float)NBINS / gm) : 0.f;
  size_t stride = (size_t)gridDim.x * blockDim.x;
  for (size_t t = (size_t)blockIdx.x * blockDim.x + threadIdx.x; t < total; t += stride) {
    int bn = (int)(S[t] * sc);
    bn = (bn > NBINS - 1) ? (NBINS - 1) : (bn < 0 ? 0 : bn);
    atomicAdd(&h[bn], 1);
  }
  __syncthreads();
  for (int i = threadIdx.x; i < NBINS; i += blockDim.x)
    if (h[i]) atomicAdd(&hist[i], h[i]);
}

__global__ void thresh_kernel(const int* hist, const int* gmaxi, float* thr) {
  float gm = __int_as_float(*gmaxi);
  int c = 0, b = NBINS - 1;
  for (; b >= 0; --b) { c += hist[b]; if (c >= TOPN) break; }
  if (b < 0) b = 0;
  *thr = gm * (float)b / (float)NBINS;
}

__global__ void compact_kernel(const float* __restrict__ S, size_t total, const float* thr,
                               int* ncand, float* candS, int* candI) {
  float t0 = *thr;
  size_t stride = (size_t)gridDim.x * blockDim.x;
  for (size_t t = (size_t)blockIdx.x * blockDim.x + threadIdx.x; t < total; t += stride) {
    float s = S[t];
    if (s >= t0) {
      int p = atomicAdd(ncand, 1);
      if (p < CAP) { candS[p] = s; candI[p] = (int)t; }
    }
  }
}

// ---------------- bitonic sort (descending score, ascending index) ----------------
__global__ void sort_kernel(float* __restrict__ ks, int* __restrict__ vs) {
  int tid = threadIdx.x;
  for (int k = 2; k <= CAP; k <<= 1) {
    for (int j = k >> 1; j > 0; j >>= 1) {
      for (int i = tid; i < CAP; i += 1024) {
        int ixj = i ^ j;
        if (ixj > i) {
          float a = ks[i], b = ks[ixj];
          int ai = vs[i], bi = vs[ixj];
          bool bBefore = (b > a) || (b == a && bi < ai);
          bool up = ((i & k) == 0);
          if (up ? bBefore : !bBefore) { ks[i] = b; ks[ixj] = a; vs[i] = bi; vs[ixj] = ai; }
        }
      }
      __syncthreads();
    }
  }
}

// ---------------- sequential NMS over sorted candidates ----------------
__global__ void nms_kernel(const int* __restrict__ candI, const float* __restrict__ srcp,
                           int n_r, int* s0, int* r0) {
  __shared__ float sx[BW], sy[BW], sz[BW];
  __shared__ int cnt, supp;
  int tid = threadIdx.x;
  if (tid < BW) {
    sx[tid] = 1e6f; sy[tid] = 1e6f; sz[tid] = 1e6f;
    int fi = candI[tid];
    s0[tid] = fi / n_r; r0[tid] = fi % n_r;   // init like reference: cs[:BW], cr[:BW]
  }
  if (tid == 0) cnt = 0;
  __syncthreads();
  for (int t = 0; t < TOPN; ++t) {
    int fi = candI[t];
    int cs = fi / n_r, cr = fi % n_r;
    float px = srcp[cs * 3], py = srcp[cs * 3 + 1], pz = srcp[cs * 3 + 2];
    if (tid == 0) supp = 0;
    __syncthreads();
    float dx = sx[tid] - px, dy = sy[tid] - py, dz = sz[tid] - pz;
    if (dx * dx + dy * dy + dz * dz < NMS_RANGE * NMS_RANGE) atomicOr(&supp, 1);
    __syncthreads();
    if (tid == 0 && !supp && cnt < BW) {
      sx[cnt] = px; sy[cnt] = py; sz[cnt] = pz;
      s0[cnt] = cs; r0[cnt] = cr; cnt++;
    }
    __syncthreads();
  }
}

// ---------------- kNN (24) + planarity + adjacency (9) ----------------
__global__ void knn_planarity_kernel(const float* __restrict__ pts, int n,
                                     int* __restrict__ adj, float* __restrict__ plan) {
  __shared__ float redv[128]; __shared__ int redi[128];
  __shared__ float lastd; __shared__ int lasti;
  __shared__ int sel[KCOV];
  int p = blockIdx.x, tid = threadIdx.x;
  float px = pts[p * 3], py = pts[p * 3 + 1], pz = pts[p * 3 + 2];
  if (tid == 0) { lastd = -1.f; lasti = -1; }
  __syncthreads();
  for (int k = 0; k < KCOV; ++k) {
    float ld = lastd; int li = lasti;
    float bv = INFINITY; int bi = 0x7fffffff;
    for (int j = tid; j < n; j += 128) {
      float d = dist3(pts + j * 3, px, py, pz);
      bool adm = (d > ld) || (d == ld && j > li);
      if (adm && (d < bv || (d == bv && j < bi))) { bv = d; bi = j; }
    }
    argmin128(bv, bi, redv, redi, tid);
    if (tid == 0) { lastd = redv[0]; lasti = redi[0]; sel[k] = redi[0]; }
    __syncthreads();
  }
  if (tid < NUM_ADJ) adj[p * NUM_ADJ + tid] = sel[tid + 1];  // drop self (rank 0)
  if (tid == 0) {
    float mx = 0, my = 0, mz = 0;
    for (int k = 0; k < KCOV; ++k) { mx += pts[sel[k] * 3]; my += pts[sel[k] * 3 + 1]; mz += pts[sel[k] * 3 + 2]; }
    mx /= KCOV; my /= KCOV; mz /= KCOV;
    float c00 = 0, c01 = 0, c02 = 0, c11 = 0, c12 = 0, c22 = 0;
    for (int k = 0; k < KCOV; ++k) {
      float dx = pts[sel[k] * 3] - mx, dy = pts[sel[k] * 3 + 1] - my, dz = pts[sel[k] * 3 + 2] - mz;
      c00 += dx * dx; c01 += dx * dy; c02 += dx * dz;
      c11 += dy * dy; c12 += dy * dz; c22 += dz * dz;
    }
    c00 /= KCOV; c01 /= KCOV; c02 /= KCOV; c11 /= KCOV; c12 /= KCOV; c22 /= KCOV;
    // closed-form symmetric 3x3 eigenvalues
    float q = (c00 + c11 + c22) / 3.f;
    float p1 = c01 * c01 + c02 * c02 + c12 * c12;
    float p2 = (c00 - q) * (c00 - q) + (c11 - q) * (c11 - q) + (c22 - q) * (c22 - q) + 2.f * p1;
    float pp = sqrtf(p2 / 6.f + 1e-30f);
    float b00 = (c00 - q) / pp, b01 = c01 / pp, b02 = c02 / pp;
    float b11 = (c11 - q) / pp, b12 = c12 / pp, b22 = (c22 - q) / pp;
    float detB = b00 * (b11 * b22 - b12 * b12) - b01 * (b01 * b22 - b12 * b02)
               + b02 * (b01 * b12 - b11 * b02);
    float r = fminf(fmaxf(detB * 0.5f, -1.f), 1.f);
    float phi = acosf(r) / 3.f;
    float e1 = q + 2.f * pp * cosf(phi);               // largest
    float e3 = q + 2.f * pp * cosf(phi + 2.0943951f);  // smallest
    float e2 = 3.f * q - e1 - e3;                      // middle
    plan[p] = 1.f - e2 / (e1 + EPS_C);
  }
}

// ---------------- triplet seeding ----------------
__global__ void triplet_kernel(const float* __restrict__ sp, const float* __restrict__ rp,
                               const int* s0a, const int* r0a, int n_s, int n_r,
                               int* i2o, int* j2o, int* i3o, int* j3o) {
  __shared__ float redv[128]; __shared__ int redi[128];
  int b = blockIdx.x, tid = threadIdx.x;
  int s0 = s0a[b], r0 = r0a[b];
  float sx = sp[s0 * 3], sy = sp[s0 * 3 + 1], sz = sp[s0 * 3 + 2];
  float rx = rp[r0 * 3], ry = rp[r0 * 3 + 1], rz = rp[r0 * 3 + 2];
  // i2
  float bv = INFINITY; int bi = 0x7fffffff;
  for (int n = tid; n < n_s; n += 128) {
    float d = dist3(sp + n * 3, sx, sy, sz);
    float f = fabsf(d - RADIUS_C) + (d < 1e-5f ? BIGF : 0.f);
    if (f < bv || (f == bv && n < bi)) { bv = f; bi = n; }
  }
  argmin128(bv, bi, redv, redi, tid);
  int i2 = redi[0];
  float d_i2 = dist3(sp + i2 * 3, sx, sy, sz);
  // j2
  bv = INFINITY; bi = 0x7fffffff;
  for (int n = tid; n < n_r; n += 128) {
    float d = dist3(rp + n * 3, rx, ry, rz);
    float f = fabsf(d - d_i2) + (d < 1e-5f ? BIGF : 0.f);
    if (f < bv || (f == bv && n < bi)) { bv = f; bi = n; }
  }
  argmin128(bv, bi, redv, redi, tid);
  int j2 = redi[0];
  // i3
  float i2x = sp[i2 * 3], i2y = sp[i2 * 3 + 1], i2z = sp[i2 * 3 + 2];
  bv = INFINITY; bi = 0x7fffffff;
  for (int n = tid; n < n_s; n += 128) {
    float d = dist3(sp + n * 3, sx, sy, sz);
    float d2 = dist3(sp + n * 3, i2x, i2y, i2z);
    float f = fabsf(d - RADIUS_C) + ((d < 1e-5f || d2 < TRI_THR) ? BIGF : 0.f);
    if (f < bv || (f == bv && n < bi)) { bv = f; bi = n; }
  }
  argmin128(bv, bi, redv, redi, tid);
  int i3 = redi[0];
  float d_i3 = dist3(sp + i3 * 3, sx, sy, sz);
  float cross_s = dist3(sp + i3 * 3, i2x, i2y, i2z);
  // j3
  float j2x = rp[j2 * 3], j2y = rp[j2 * 3 + 1], j2z = rp[j2 * 3 + 2];
  bv = INFINITY; bi = 0x7fffffff;
  for (int n = tid; n < n_r; n += 128) {
    float d = dist3(rp + n * 3, rx, ry, rz);
    float dj2 = dist3(rp + n * 3, j2x, j2y, j2z);
    float f = fabsf(d - d_i3) + fabsf(dj2 - cross_s) + (d < 1e-5f ? BIGF : 0.f);
    if (f < bv || (f == bv && n < bi)) { bv = f; bi = n; }
  }
  argmin128(bv, bi, redv, redi, tid);
  if (tid == 0) { i2o[b] = i2; j2o[b] = j2; i3o[b] = i3; j3o[b] = redi[0]; }
}

// ---------------- weighted Procrustes via Horn quaternion (thread 0) ----------------
__device__ void procrustes_t0(const float* tsx, const float* tsy, const float* tsz,
                              const float* trx, const float* try_, const float* trz,
                              const float* w, float* T) {
  float wsum = 1e-8f;
  for (int l = 0; l < LL; ++l) wsum += w[l];
  float inv = 1.f / wsum;
  float scx = 0, scy = 0, scz = 0, rcx = 0, rcy = 0, rcz = 0;
  for (int l = 0; l < LL; ++l) {
    float wn = w[l] * inv;
    scx += wn * tsx[l]; scy += wn * tsy[l]; scz += wn * tsz[l];
    rcx += wn * trx[l]; rcy += wn * try_[l]; rcz += wn * trz[l];
  }
  float H[9] = {0, 0, 0, 0, 0, 0, 0, 0, 0};
  for (int l = 0; l < LL; ++l) {
    float wn = w[l] * inv;
    float ax = tsx[l] - scx, ay = tsy[l] - scy, az = tsz[l] - scz;
    float bx = trx[l] - rcx, by = try_[l] - rcy, bz = trz[l] - rcz;
    H[0] += wn * ax * bx; H[1] += wn * ax * by; H[2] += wn * ax * bz;
    H[3] += wn * ay * bx; H[4] += wn * ay * by; H[5] += wn * ay * bz;
    H[6] += wn * az * bx; H[7] += wn * az * by; H[8] += wn * az * bz;
  }
  float trH = H[0] + H[4] + H[8];
  float N[4][4];
  N[0][0] = trH;
  N[0][1] = H[5] - H[7]; N[0][2] = H[6] - H[2]; N[0][3] = H[1] - H[3];
  N[1][0] = N[0][1]; N[2][0] = N[0][2]; N[3][0] = N[0][3];
  N[1][1] = 2.f * H[0] - trH; N[2][2] = 2.f * H[4] - trH; N[3][3] = 2.f * H[8] - trH;
  N[1][2] = H[1] + H[3]; N[2][1] = N[1][2];
  N[1][3] = H[2] + H[6]; N[3][1] = N[1][3];
  N[2][3] = H[5] + H[7]; N[3][2] = N[2][3];
  float fr = 0.f;
  for (int i = 0; i < 4; ++i) for (int j = 0; j < 4; ++j) fr += N[i][j] * N[i][j];
  fr = sqrtf(fr) + 1e-6f;   // shift >= spectral radius -> power iter finds max eigenvalue
  float q0 = 1.f, q1 = 0.f, q2 = 0.f, q3 = 0.f;
  for (int it = 0; it < 64; ++it) {
    float a0 = (N[0][0] + fr) * q0 + N[0][1] * q1 + N[0][2] * q2 + N[0][3] * q3;
    float a1 = N[1][0] * q0 + (N[1][1] + fr) * q1 + N[1][2] * q2 + N[1][3] * q3;
    float a2 = N[2][0] * q0 + N[2][1] * q1 + (N[2][2] + fr) * q2 + N[2][3] * q3;
    float a3 = N[3][0] * q0 + N[3][1] * q1 + N[3][2] * q2 + (N[3][3] + fr) * q3;
    float nn = rsqrtf(a0 * a0 + a1 * a1 + a2 * a2 + a3 * a3 + 1e-30f);
    q0 = a0 * nn; q1 = a1 * nn; q2 = a2 * nn; q3 = a3 * nn;
  }
  float R00 = q0 * q0 + q1 * q1 - q2 * q2 - q3 * q3;
  float R01 = 2.f * (q1 * q2 - q0 * q3);
  float R02 = 2.f * (q1 * q3 + q0 * q2);
  float R10 = 2.f * (q1 * q2 + q0 * q3);
  float R11 = q0 * q0 - q1 * q1 + q2 * q2 - q3 * q3;
  float R12 = 2.f * (q2 * q3 - q0 * q1);
  float R20 = 2.f * (q1 * q3 - q0 * q2);
  float R21 = 2.f * (q2 * q3 + q0 * q1);
  float R22 = q0 * q0 - q1 * q1 - q2 * q2 + q3 * q3;
  T[0] = R00; T[1] = R01; T[2] = R02;
  T[3] = R10; T[4] = R11; T[5] = R12;
  T[6] = R20; T[7] = R21; T[8] = R22;
  T[9]  = rcx - (R00 * scx + R01 * scy + R02 * scz);
  T[10] = rcy - (R10 * scx + R11 * scy + R12 * scz);
  T[11] = rcz - (R20 * scx + R21 * scy + R22 * scz);
}

// ---------------- flood-fill main loop: one block per batch element ----------------
__global__ void __launch_bounds__(128) mainloop_kernel(
    const float* __restrict__ sp, const float* __restrict__ rp,
    const int* __restrict__ adjS, const int* __restrict__ adjR,
    const int* s0a, const int* r0a, const int* i2a, const int* j2a,
    const int* i3a, const int* j3a, float* __restrict__ Tout, int* __restrict__ inl) {
  __shared__ int nodeS[LL], nodeR[LL];
  __shared__ float tsx[LL], tsy[LL], tsz[LL], trx[LL], try_[LL], trz[LL];
  __shared__ float mw[LL], wsh[LL];
  __shared__ unsigned usb[128], urb[128];          // 4096-point used bitmasks
  __shared__ float sgx[NSTAGE], sgy[NSTAGE], sgz[NSTAGE], svs[NSTAGE];
  __shared__ int sni[NSTAGE];
  __shared__ float rgx[NSTAGE], rgy[NSTAGE], rgz[NSTAGE], svr[NSTAGE];
  __shared__ int rni[NSTAGE];
  __shared__ float redv[128]; __shared__ int redi[128];
  __shared__ float T[12];
  int b = blockIdx.x, tid = threadIdx.x;

  for (int l = tid; l < LL; l += 128) { nodeS[l] = 0; nodeR[l] = 0; mw[l] = 1.f; }
  usb[tid] = 0; urb[tid] = 0;
  __syncthreads();
  if (tid == 0) {
    nodeS[0] = s0a[b]; nodeR[0] = r0a[b];
    nodeS[1] = i2a[b]; nodeR[1] = j2a[b];
    nodeS[2] = i3a[b]; nodeR[2] = j3a[b];
    for (int k = 0; k < 3; ++k) {
      usb[nodeS[k] >> 5] |= 1u << (nodeS[k] & 31);
      urb[nodeR[k] >> 5] |= 1u << (nodeR[k] & 31);
    }
  }
  __syncthreads();
  for (int l = tid; l < LL; l += 128) {
    int si = nodeS[l], ri = nodeR[l];
    tsx[l] = sp[si * 3]; tsy[l] = sp[si * 3 + 1]; tsz[l] = sp[si * 3 + 2];
    trx[l] = rp[ri * 3]; try_[l] = rp[ri * 3 + 1]; trz[l] = rp[ri * 3 + 2];
    wsh[l] = (l < 3) ? 1.f : 0.f;
  }
  __syncthreads();
  if (tid == 0) procrustes_t0(tsx, tsy, tsz, trx, try_, trz, wsh, T);
  __syncthreads();

  for (int it = 0; it < FSTP; ++it) {
    int cur = 3 + it;
    // stage all (node, neighbor) entries
    for (int e = tid; e < NSTAGE; e += 128) {
      int l = e / NUM_ADJ, k = e % NUM_ADJ;
      int si = adjS[nodeS[l] * NUM_ADJ + k];
      float px = sp[si * 3], py = sp[si * 3 + 1], pz = sp[si * 3 + 2];
      sgx[e] = T[0] * px + T[1] * py + T[2] * pz + T[9];
      sgy[e] = T[3] * px + T[4] * py + T[5] * pz + T[10];
      sgz[e] = T[6] * px + T[7] * py + T[8] * pz + T[11];
      svs[e] = ((usb[si >> 5] >> (si & 31)) & 1u) ? 100.f : 0.f;
      sni[e] = si;
      int ri = adjR[nodeR[l] * NUM_ADJ + k];
      rgx[e] = rp[ri * 3]; rgy[e] = rp[ri * 3 + 1]; rgz[e] = rp[ri * 3 + 2];
      svr[e] = ((urb[ri >> 5] >> (ri & 31)) & 1u) ? 100.f : 0.f;
      rni[e] = ri;
    }
    __syncthreads();
    // argmin over (l, i, j)
    float bv = INFINITY; int bi = 0;
    for (int c = tid; c < LL * NUM_ADJ * NUM_ADJ; c += 128) {
      int l = c / 81, ii = (c % 81) / 9, jj = c % 9;
      int eS = l * NUM_ADJ + ii, eR = l * NUM_ADJ + jj;
      float dx = sgx[eS] - rgx[eR], dy = sgy[eS] - rgy[eR], dz = sgz[eS] - rgz[eR];
      float v = sqrtf(dx * dx + dy * dy + dz * dz + EPS_C) + svs[eS] + svr[eR]
              + (l < cur ? 0.f : 1e9f);
      if (v < bv) { bv = v; bi = c; }
    }
    argmin128(bv, bi, redv, redi, tid);
    int best = redi[0];
    if (tid == 0) {
      int l = best / 81, ii = (best % 81) / 9, jj = best % 9;
      int ns = sni[l * NUM_ADJ + ii], nr = rni[l * NUM_ADJ + jj];
      nodeS[cur] = ns; nodeR[cur] = nr;
      usb[ns >> 5] |= 1u << (ns & 31);
      urb[nr >> 5] |= 1u << (nr & 31);
      tsx[cur] = sp[ns * 3]; tsy[cur] = sp[ns * 3 + 1]; tsz[cur] = sp[ns * 3 + 2];
      trx[cur] = rp[nr * 3]; try_[cur] = rp[nr * 3 + 1]; trz[cur] = rp[nr * 3 + 2];
    }
    __syncthreads();
    for (int l = tid; l < LL; l += 128) wsh[l] = mw[l] * (l <= cur ? 1.f : 0.f);
    __syncthreads();
    if (tid == 0) procrustes_t0(tsx, tsy, tsz, trx, try_, trz, wsh, T);
    __syncthreads();
    for (int l = tid; l < LL; l += 128) {
      if (l <= cur) {
        float ex = T[0] * tsx[l] + T[1] * tsy[l] + T[2] * tsz[l] + T[9] - trx[l];
        float ey = T[3] * tsx[l] + T[4] * tsy[l] + T[5] * tsz[l] + T[10] - try_[l];
        float ez = T[6] * tsx[l] + T[7] * tsy[l] + T[8] * tsz[l] + T[11] - trz[l];
        float res = sqrtf(ex * ex + ey * ey + ez * ez + EPS_C);
        mw[l] = fmaxf(1.f - res * res / (INLIER_C * INLIER_C), 0.f);
      }
    }
    __syncthreads();
  }

  // LGR refinement
  for (int g = 0; g < LGR; ++g) {
    for (int l = tid; l < LL; l += 128) {
      float ex = T[0] * tsx[l] + T[1] * tsy[l] + T[2] * tsz[l] + T[9] - trx[l];
      float ey = T[3] * tsx[l] + T[4] * tsy[l] + T[5] * tsz[l] + T[10] - try_[l];
      float ez = T[6] * tsx[l] + T[7] * tsy[l] + T[8] * tsz[l] + T[11] - trz[l];
      float res = sqrtf(ex * ex + ey * ey + ez * ez + EPS_C);
      wsh[l] = (res < INLIER_C ? 1.f : 0.f) + 1e-6f;
    }
    __syncthreads();
    if (tid == 0) procrustes_t0(tsx, tsy, tsz, trx, try_, trz, wsh, T);
    __syncthreads();
  }
  // inlier count
  int cnt = 0;
  for (int l = tid; l < LL; l += 128) {
    float ex = T[0] * tsx[l] + T[1] * tsy[l] + T[2] * tsz[l] + T[9] - trx[l];
    float ey = T[3] * tsx[l] + T[4] * tsy[l] + T[5] * tsz[l] + T[10] - try_[l];
    float ez = T[6] * tsx[l] + T[7] * tsy[l] + T[8] * tsz[l] + T[11] - trz[l];
    if (sqrtf(ex * ex + ey * ey + ez * ez + EPS_C) < INLIER_C) cnt++;
  }
  redi[tid] = cnt; __syncthreads();
  for (int s = 64; s > 0; s >>= 1) { if (tid < s) redi[tid] += redi[tid + s]; __syncthreads(); }
  if (tid == 0) {
    inl[b] = redi[0];
    for (int k = 0; k < 12; ++k) Tout[b * 12 + k] = T[k];
  }
}

// ---------------- pick best batch element, emit 4x4 ----------------
__global__ void select_kernel(const float* __restrict__ Tout, const int* __restrict__ inl,
                              float* __restrict__ out) {
  __shared__ int rv[BW], ri[BW];
  int tid = threadIdx.x;
  rv[tid] = inl[tid]; ri[tid] = tid;
  __syncthreads();
  for (int s = BW / 2; s > 0; s >>= 1) {
    if (tid < s) {
      int a = rv[tid], b = rv[tid + s], ai = ri[tid], bi = ri[tid + s];
      if (b > a || (b == a && bi < ai)) { rv[tid] = b; ri[tid] = bi; }
    }
    __syncthreads();
  }
  if (tid == 0) {
    const float* T = Tout + ri[0] * 12;
    out[0] = T[0];  out[1] = T[1];  out[2]  = T[2];  out[3]  = T[9];
    out[4] = T[3];  out[5] = T[4];  out[6]  = T[5];  out[7]  = T[10];
    out[8] = T[6];  out[9] = T[7];  out[10] = T[8];  out[11] = T[11];
    out[12] = 0.f;  out[13] = 0.f;  out[14] = 0.f;   out[15] = 1.f;
  }
}

// ---------------- host side ----------------
extern "C" void kernel_launch(void* const* d_in, const int* in_sizes, int n_in,
                              void* d_out, int out_size, void* d_ws, size_t ws_size,
                              hipStream_t stream) {
  const float* srcp  = (const float*)d_in[0];
  const float* refp  = (const float*)d_in[1];
  const float* sfeat = (const float*)d_in[2];
  const float* rfeat = (const float*)d_in[3];
  int n_s = in_sizes[0] / 3;
  int n_r = in_sizes[1] / 3;
  int D   = in_sizes[2] / n_s;
  size_t total = (size_t)n_s * n_r;

  char* base = (char*)d_ws; size_t off = 0;
  auto carve = [&](size_t bytes) -> char* {
    char* p = base + off; off = (off + bytes + 255) & ~(size_t)255; return p;
  };
  bf16_t* sbf   = (bf16_t*)carve((size_t)n_s * D * 2);
  bf16_t* rbf   = (bf16_t*)carve((size_t)n_r * D * 2);
  float*  score = (float*) carve(total * 4);           // logits, then score in place
  float*  rmax  = (float*) carve((size_t)n_s * 4);
  float*  rsum  = (float*) carve((size_t)n_s * 4);
  float*  cmax  = (float*) carve((size_t)n_r * 4);
  float*  csum  = (float*) carve((size_t)n_r * 4);
  float*  es    = (float*) carve((size_t)n_s * 4);
  float*  er    = (float*) carve((size_t)n_r * 4);
  int*    adjS  = (int*)   carve((size_t)n_s * NUM_ADJ * 4);
  int*    adjR  = (int*)   carve((size_t)n_r * NUM_ADJ * 4);
  int*    gmax  = (int*)   carve(4);
  int*    ncand = (int*)   carve(4);
  float*  thr   = (float*) carve(4);
  int*    hist  = (int*)   carve((size_t)NBINS * 4);
  float*  candS = (float*) carve((size_t)CAP * 4);
  int*    candI = (int*)   carve((size_t)CAP * 4);
  int*    s0    = (int*)   carve(BW * 4);
  int*    r0    = (int*)   carve(BW * 4);
  int*    i2    = (int*)   carve(BW * 4);
  int*    j2    = (int*)   carve(BW * 4);
  int*    i3    = (int*)   carve(BW * 4);
  int*    j3    = (int*)   carve(BW * 4);
  float*  Tout  = (float*) carve((size_t)BW * 12 * 4);
  int*    inl   = (int*)   carve(BW * 4);

  init_kernel<<<(CAP + 255) / 256, 256, 0, stream>>>(gmax, ncand, hist, candS, candI);

  knn_planarity_kernel<<<n_s, 128, 0, stream>>>(srcp, n_s, adjS, es);
  knn_planarity_kernel<<<n_r, 128, 0, stream>>>(refp, n_r, adjR, er);

  cvt_bf16_kernel<<<((n_s * D) + 255) / 256, 256, 0, stream>>>(sfeat, sbf, n_s * D);
  cvt_bf16_kernel<<<((n_r * D) + 255) / 256, 256, 0, stream>>>(rfeat, rbf, n_r * D);

  {
    dim3 blk(32, 4);
    dim3 grd(n_r / 64, n_s / 64);
    float scale = 1.0f / sqrtf((float)D);
    wmma_logits_kernel<<<grd, blk, 0, stream>>>(sbf, rbf, score, n_r, D, scale);
  }

  row_stats_kernel<<<n_s, 256, 0, stream>>>(score, rmax, rsum, n_r);
  col_stats_kernel<<<n_r, 256, 0, stream>>>(score, cmax, csum, n_s, n_r);
  score_kernel<<<4096, 256, 0, stream>>>(score, rmax, rsum, cmax, csum, es, er, n_s, n_r, gmax);

  hist_kernel<<<2048, 256, 0, stream>>>(score, total, gmax, hist);
  thresh_kernel<<<1, 1, 0, stream>>>(hist, gmax, thr);
  compact_kernel<<<4096, 256, 0, stream>>>(score, total, thr, ncand, candS, candI);
  sort_kernel<<<1, 1024, 0, stream>>>(candS, candI);
  nms_kernel<<<1, BW, 0, stream>>>(candI, srcp, n_r, s0, r0);

  triplet_kernel<<<BW, 128, 0, stream>>>(srcp, refp, s0, r0, n_s, n_r, i2, j2, i3, j3);
  mainloop_kernel<<<BW, 128, 0, stream>>>(srcp, refp, adjS, adjR, s0, r0, i2, j2, i3, j3,
                                          Tout, inl);
  select_kernel<<<1, BW, 0, stream>>>(Tout, inl, (float*)d_out);
}
